// Net_6837587935245
// MI455X (gfx1250) — matrix operand order repaired and checked by reading, not compiled, not run on hardware
//
#include <hip/hip_runtime.h>
#include <hip/hip_bf16.h>
#include <math.h>
#include <stdint.h>

#define N_PTS 32768
#define NPB   2048     // points per batch (N/B)
#define KNN   30
#define DILF  8
#define BN_EPS 1e-5f

typedef __attribute__((ext_vector_type(16))) _Float16 v16h;
typedef __attribute__((ext_vector_type(8)))  float    v8f;
typedef __attribute__((ext_vector_type(4)))  unsigned su4;
typedef __attribute__((ext_vector_type(8)))  unsigned su8;

// ---------------- WMMA fragment helpers (CDNA5 layouts, CDNA5 ISA) ----------------

// 16-bit A matrix 16x32 (MxK): lane L -> row m = L%16, h = L/16
// elems 0..7  : K = h*8 + 0..7 ; elems 8..15 : K = 16 + h*8 + 0..7
__device__ inline v16h load_A_frag(const _Float16* A, int lda, int row0, int k0, int lane) {
  int m = lane & 15, h = lane >> 4;
  const _Float16* p = A + (size_t)(row0 + m) * lda + k0;
  v16h a;
#pragma unroll
  for (int j = 0; j < 8; ++j) a[j] = p[h * 8 + j];
#pragma unroll
  for (int j = 0; j < 8; ++j) a[8 + j] = p[16 + h * 8 + j];
  return a;
}

// Same A-fragment layout but sourced from an LDS tile with padded row stride.
__device__ inline v16h lds_A_frag(const _Float16* sA, int ldlds, int rowl, int ksub, int lane) {
  int m = lane & 15, h = lane >> 4;
  const _Float16* p = sA + (rowl + m) * ldlds + ksub;
  v16h a;
#pragma unroll
  for (int j = 0; j < 8; ++j) a[j] = p[h * 8 + j];
#pragma unroll
  for (int j = 0; j < 8; ++j) a[8 + j] = p[16 + h * 8 + j];
  return a;
}

// 16-bit B matrix 32x16 (KxN): lane L -> col n = L%16, h = L/16
// elems 0..7 : K = h*8 + 0..7 ; elems 8..15 : K = 16 + h*8 + 0..7
// Wt is stored N-major: Wt[n*K + k]
__device__ inline v16h load_B_frag(const _Float16* Wt, int K, int n0, int k0, int lane) {
  int n = n0 + (lane & 15), h = lane >> 4;
  const _Float16* p = Wt + (size_t)n * K + k0;
  v16h b;
#pragma unroll
  for (int j = 0; j < 8; ++j) b[j] = p[h * 8 + j];
#pragma unroll
  for (int j = 0; j < 8; ++j) b[8 + j] = p[16 + h * 8 + j];
  return b;
}

__device__ inline v8f wmma_f16(v16h a, v16h b, v8f c) {
  return __builtin_amdgcn_wmma_f32_16x16x32_f16(false, a, false, b, (short)0, c, false, false);
}

__device__ inline void atomic_max_f32(float* addr, float v) {
  __hip_atomic_fetch_max(addr, v, __ATOMIC_RELAXED, __HIP_MEMORY_SCOPE_AGENT);
}

// ---------------- Tensor Data Mover: 2D tile (64 rows x 64 halves) global -> LDS ----------------
// D# per CDNA5 ISA. data_size=2B. LDS padding: after every
// 32 dwords (one 64-half row) insert 4 dwords -> LDS row stride 72 halves (bank-conflict-free).
__device__ inline void tdm_load_tile64x64(const _Float16* gsrc, unsigned lds_off, unsigned kin) {
  unsigned long long ga = (unsigned long long)(uintptr_t)gsrc;
  su4 g0;
  g0[0] = 1u;                                   // count=1, user-mode descriptor
  g0[1] = lds_off;                              // lds_addr (bytes)
  g0[2] = (unsigned)(ga & 0xFFFFFFFFu);         // global_addr[31:0]
  g0[3] = (unsigned)((ga >> 32) & 0x01FFFFFFu)  // global_addr[56:32]
          | (2u << 30);                         // type = 2 ("image")
  su8 g1;
  unsigned td0 = kin;                            // tensor_dim0 (elements, OOB bound)
  unsigned td1 = 1u << 20;                       // tensor_dim1 (rows, generous bound)
  g1[0] = (1u << 16)                             // data_size = 2 bytes
        | (1u << 20)                             // pad_enable
        | (4u << 22)                             // pad_interval: 32 dwords
        | (3u << 25);                            // pad_amount: 4 dwords
  g1[1] = (td0 & 0xFFFFu) << 16;                 // tensor_dim0[15:0]
  g1[2] = (td0 >> 16) | ((td1 & 0xFFFFu) << 16); // tensor_dim0[31:16] | tensor_dim1[15:0]
  g1[3] = (td1 >> 16) | (64u << 16);             // tensor_dim1[31:16] | tile_dim0 = 64
  g1[4] = 64u;                                   // tile_dim1 = 64 rows, tile_dim2 = 0
  g1[5] = kin;                                   // tensor_dim0_stride[31:0] (elements)
  g1[6] = 0u;                                    // stride hi | tensor_dim1_stride lo
  g1[7] = 0u;
  asm volatile("tensor_load_to_lds %0, %1" :: "s"(g0), "s"(g1) : "memory");
}

// ---------------- utility kernels ----------------

__global__ void k_fill_f32(float* p, float v, int count) {
  int t = blockIdx.x * blockDim.x + threadIdx.x;
  if (t < count) p[t] = v;
}

// src: f32 row-major (K x Nn) -> dst: f16 N-major (Nn rows of K contiguous)
__global__ void k_transpose_f32_to_f16(const float* src, _Float16* dst, int K, int Nn) {
  int t = blockIdx.x * blockDim.x + threadIdx.x;
  if (t >= K * Nn) return;
  int k = t % K, n = t / K;
  dst[(size_t)n * K + k] = (_Float16)src[(size_t)k * Nn + n];
}

// ---------------- kNN (per-batch brute force + top-30) ----------------

template <int D>
__global__ void k_knn(const float* __restrict__ feat, int* __restrict__ idx_out) {
  __shared__ float tile[256 * D];
  int tid = threadIdx.x;
  int q = blockIdx.x * 256 + tid;
  int b = q / NPB;
  int base = b * NPB;

  float xq[D];
#pragma unroll
  for (int d = 0; d < D; ++d) xq[d] = feat[(size_t)q * D + d];

  float best[KNN];
  int   bidx[KNN];
  int cnt = 0, wpos = 0;
  float worst = -1.0f;

  for (int t0 = 0; t0 < NPB; t0 += 256) {
    __syncthreads();
#pragma unroll
    for (int d = 0; d < D; ++d) tile[tid * D + d] = feat[(size_t)(base + t0 + tid) * D + d];
    __syncthreads();
    for (int c = 0; c < 256; ++c) {
      float dist = 0.0f;
#pragma unroll
      for (int d = 0; d < D; ++d) {
        float df = xq[d] - tile[c * D + d];
        dist += df * df;
      }
      int j = base + t0 + c;
      if (cnt < KNN) {
        best[cnt] = dist; bidx[cnt] = j;
        if (dist > worst) { worst = dist; wpos = cnt; }
        ++cnt;
      } else if (dist < worst) {
        best[wpos] = dist; bidx[wpos] = j;
        worst = best[0]; wpos = 0;
        for (int u = 1; u < KNN; ++u)
          if (best[u] > worst) { worst = best[u]; wpos = u; }
      }
    }
  }
  for (int i = 1; i < KNN; ++i) {
    float dv = best[i]; int iv = bidx[i]; int j2 = i - 1;
    while (j2 >= 0 && best[j2] > dv) { best[j2 + 1] = best[j2]; bidx[j2 + 1] = bidx[j2]; --j2; }
    best[j2 + 1] = dv; bidx[j2 + 1] = iv;
  }
  for (int i = 0; i < KNN; ++i) idx_out[(size_t)q * KNN + i] = bidx[i];
}

// ---------------- EdgeConv layer 1: gather + (fin->64) + relu + BN, f16 out ----------------

template <int D>  // feature dim; fin = 2*D
__global__ void k_edge_layer1(const float* __restrict__ feat, const int* __restrict__ idx,
                              int dil, int nEdges,
                              const float* __restrict__ W1, const float* __restrict__ b1,
                              const float* __restrict__ beta, const float* __restrict__ g,
                              const float* __restrict__ mu, const float* __restrict__ var,
                              _Float16* __restrict__ e1) {
  constexpr int FIN = 2 * D;
  __shared__ float sW[FIN * 64];
  __shared__ float sB[64], sA[64], sC[64];
  int tid = threadIdx.x;
  for (int t = tid; t < FIN * 64; t += blockDim.x) sW[t] = W1[t];
  if (tid < 64) {
    float a = g[tid] * rsqrtf(var[tid] + BN_EPS);
    sB[tid] = b1[tid];
    sA[tid] = a;
    sC[tid] = beta[tid] - mu[tid] * a;
  }
  __syncthreads();

  int e = blockIdx.x * blockDim.x + tid;
  if (e >= nEdges) return;
  int f   = e * dil;
  int tgt = f / KNN;
  int src = idx[f];

  float h[FIN];
#pragma unroll
  for (int d = 0; d < D; ++d) {
    float xi = feat[(size_t)tgt * D + d];
    float xj = feat[(size_t)src * D + d];
    h[d] = xi;
    h[D + d] = xj - xi;
  }
  for (int c = 0; c < 64; ++c) {
    float z = sB[c];
#pragma unroll
    for (int d = 0; d < FIN; ++d) z += h[d] * sW[d * 64 + c];
    z = z > 0.0f ? z : 0.0f;
    e1[(size_t)e * 64 + c] = (_Float16)(z * sA[c] + sC[c]);
  }
}

// ---------------- EdgeConv layer 2: (E x 64) @ (64 x 64) via WMMA + BN + scatter-max ----------------

__global__ void k_edge_layer2_wmma(const _Float16* __restrict__ e1, const _Float16* __restrict__ W2t,
                                   const float* __restrict__ b2, const float* __restrict__ beta,
                                   const float* __restrict__ g, const float* __restrict__ mu,
                                   const float* __restrict__ var,
                                   int dil, int nEdges, float* __restrict__ out /* N x 64 */) {
  int wave = threadIdx.x >> 5;
  int lane = threadIdx.x & 31;
  int row0 = blockIdx.x * 64 + wave * 16;

  v8f acc[4] = {v8f{}, v8f{}, v8f{}, v8f{}};
  v16h a0 = load_A_frag(e1, 64, row0, 0, lane);
  v16h a1 = load_A_frag(e1, 64, row0, 32, lane);
#pragma unroll
  for (int nt = 0; nt < 4; ++nt) {
    v16h b0 = load_B_frag(W2t, 64, nt * 16, 0, lane);
    v16h b1 = load_B_frag(W2t, 64, nt * 16, 32, lane);
    acc[nt] = wmma_f16(a0, b0, acc[nt]);
    acc[nt] = wmma_f16(a1, b1, acc[nt]);
  }

  int h = lane >> 4, nl = lane & 15;
#pragma unroll
  for (int nt = 0; nt < 4; ++nt) {
    int ch = nt * 16 + nl;
    float aa = g[ch] * rsqrtf(var[ch] + BN_EPS);
    float cc = beta[ch] - mu[ch] * aa;
    float bb = b2[ch];
#pragma unroll
    for (int r = 0; r < 8; ++r) {
      int edge = row0 + r + 8 * h;
      if (edge >= nEdges) continue;
      float z = acc[nt][r] + bb;
      z = z > 0.0f ? z : 0.0f;
      float y = z * aa + cc;
      int tgt = (edge * dil) / KNN;
      atomic_max_f32(&out[(size_t)tgt * 64 + ch], y);
    }
  }
}

// ---------------- concat [x1|x2|x3|x4] -> f16 (N x 256) ----------------

__global__ void k_concat4_f16(const float* __restrict__ x1, const float* __restrict__ x2,
                              const float* __restrict__ x3, const float* __restrict__ x4,
                              _Float16* __restrict__ h) {
  int t = blockIdx.x * blockDim.x + threadIdx.x;
  if (t >= N_PTS * 256) return;
  int i = t >> 8, ch = t & 255;
  const float* src = (ch < 64) ? x1 : (ch < 128) ? x2 : (ch < 192) ? x3 : x4;
  h[t] = (_Float16)src[(size_t)i * 64 + (ch & 63)];
}

// ---------------- dense MLP block via WMMA with TDM-staged A tiles ----------------
// Block = 128 threads (4 waves). Block tile = 64 rows x 64 cols. The 64x64-half A tile
// is DMA'd into LDS by the Tensor Data Mover (with padded rows: stride 72 halves),
// completion tracked with TENSORcnt; fragments then come from LDS.

#define LDS_LDA 72

__global__ void k_lin_wmma(const _Float16* __restrict__ A, const _Float16* __restrict__ Wt,
                           int Kin, int Nout,
                           const float* __restrict__ bias, const float* __restrict__ beta,
                           const float* __restrict__ g, const float* __restrict__ mu,
                           const float* __restrict__ var, _Float16* __restrict__ out) {
  __shared__ _Float16 sA[64 * LDS_LDA];
  int wave = threadIdx.x >> 5;
  int lane = threadIdx.x & 31;
  int blockRow0 = blockIdx.x * 64;
  int row0 = blockRow0 + wave * 16;
  int col0 = blockIdx.y * 64;

  v8f acc[4] = {v8f{}, v8f{}, v8f{}, v8f{}};
  for (int k0 = 0; k0 < Kin; k0 += 64) {
    __syncthreads();  // previous chunk's consumers done before tile overwrite
    if (threadIdx.x == 0) {
      tdm_load_tile64x64(A + (size_t)blockRow0 * Kin + k0,
                         (unsigned)(uintptr_t)sA, (unsigned)Kin);
      __builtin_amdgcn_s_wait_tensorcnt(0);
    }
    __syncthreads();  // tile visible to all waves

    v16h a0 = lds_A_frag(sA, LDS_LDA, wave * 16, 0, lane);
    v16h a1 = lds_A_frag(sA, LDS_LDA, wave * 16, 32, lane);
#pragma unroll
    for (int nt = 0; nt < 4; ++nt) {
      v16h b0 = load_B_frag(Wt, Kin, col0 + nt * 16, k0, lane);
      v16h b1 = load_B_frag(Wt, Kin, col0 + nt * 16, k0 + 32, lane);
      acc[nt] = wmma_f16(a0, b0, acc[nt]);
      acc[nt] = wmma_f16(a1, b1, acc[nt]);
    }
  }

  int h = lane >> 4, nl = lane & 15;
#pragma unroll
  for (int nt = 0; nt < 4; ++nt) {
    int ch = col0 + nt * 16 + nl;
    float aa = g[ch] * rsqrtf(var[ch] + BN_EPS);
    float cc = beta[ch] - mu[ch] * aa;
    float bb = bias[ch];
#pragma unroll
    for (int r = 0; r < 8; ++r) {
      int row = row0 + r + 8 * h;
      float z = acc[nt][r] + bb;
      z = z > 0.0f ? z : 0.0f;
      out[(size_t)row * Nout + ch] = (_Float16)(z * aa + cc);
    }
  }
}

// ---------------- final 128->10 + log_softmax (f32 out) ----------------

__global__ void k_lout_logsoftmax(const _Float16* __restrict__ h3, const float* __restrict__ W,
                                  const float* __restrict__ b, float* __restrict__ out) {
  int i = blockIdx.x * blockDim.x + threadIdx.x;
  if (i >= N_PTS) return;
  const _Float16* hr = h3 + (size_t)i * 128;
  float logits[10];
  float m = -INFINITY;
#pragma unroll
  for (int c = 0; c < 10; ++c) {
    float z = b[c];
    for (int k = 0; k < 128; ++k) z += (float)hr[k] * W[k * 10 + c];
    logits[c] = z;
    m = fmaxf(m, z);
  }
  float s = 0.0f;
#pragma unroll
  for (int c = 0; c < 10; ++c) s += expf(logits[c] - m);
  float lse = m + logf(s);
#pragma unroll
  for (int c = 0; c < 10; ++c) out[(size_t)i * 10 + c] = logits[c] - lse;
}

// ---------------- host orchestration ----------------

extern "C" void kernel_launch(void* const* d_in, const int* in_sizes, int n_in,
                              void* d_out, int out_size, void* d_ws, size_t ws_size,
                              hipStream_t stream) {
  (void)in_sizes; (void)n_in; (void)out_size; (void)ws_size;

  auto P = [&](int i) { return (const float*)d_in[i]; };
  const float* x   = P(0);   // (N, 25)
  const float* pos = P(1);   // (N, 3)
  // params flattened (dict keys sorted): conv1..conv4 (2 blocks, fields W,b,beta,g,mu,var),
  // lin1 (1 block), lout (W,b), m1, m2
  const int C1A = 3,  C1B = 9,  C2A = 15, C2B = 21;
  const int C3A = 27, C3B = 33, C4A = 39, C4B = 45;
  const int L1 = 51, LOUT_W = 57, LOUT_B = 58, M1 = 59, M2 = 65;

  char* ws = (char*)d_ws;
  size_t off = 0;
  auto alloc = [&](size_t bytes) -> void* {
    off = (off + 255) & ~(size_t)255;
    void* p = ws + off;
    off += bytes;
    return p;
  };
  const int E_full = N_PTS * KNN;       // 983040
  const int E_dil  = E_full / DILF;     // 122880

  int*      idx_pos = (int*)alloc((size_t)N_PTS * KNN * 4);
  int*      idx_x   = (int*)alloc((size_t)N_PTS * KNN * 4);
  _Float16* e1      = (_Float16*)alloc((size_t)E_full * 64 * 2);
  float*    x1      = (float*)alloc((size_t)N_PTS * 64 * 4);
  float*    x2      = (float*)alloc((size_t)N_PTS * 64 * 4);
  float*    x3      = (float*)alloc((size_t)N_PTS * 64 * 4);
  float*    x4      = (float*)alloc((size_t)N_PTS * 64 * 4);
  _Float16* hcat    = (_Float16*)alloc((size_t)N_PTS * 256 * 2);
  _Float16* h1      = (_Float16*)alloc((size_t)N_PTS * 1024 * 2);
  _Float16* h2      = (_Float16*)alloc((size_t)N_PTS * 256 * 2);
  _Float16* h3      = (_Float16*)alloc((size_t)N_PTS * 128 * 2);
  _Float16* w2t[4];
  for (int c = 0; c < 4; ++c) w2t[c] = (_Float16*)alloc((size_t)64 * 64 * 2);
  _Float16* lin1t = (_Float16*)alloc((size_t)1024 * 256 * 2);
  _Float16* m1t   = (_Float16*)alloc((size_t)256 * 1024 * 2);
  _Float16* m2t   = (_Float16*)alloc((size_t)128 * 256 * 2);

  // 1) weight prep
  {
    int cw[4] = {C1B, C2B, C3B, C4B};
    for (int c = 0; c < 4; ++c)
      k_transpose_f32_to_f16<<<(64 * 64 + 255) / 256, 256, 0, stream>>>(P(cw[c]), w2t[c], 64, 64);
    k_transpose_f32_to_f16<<<(256 * 1024 + 255) / 256, 256, 0, stream>>>(P(L1), lin1t, 256, 1024);
    k_transpose_f32_to_f16<<<(1024 * 256 + 255) / 256, 256, 0, stream>>>(P(M1), m1t, 1024, 256);
    k_transpose_f32_to_f16<<<(256 * 128 + 255) / 256, 256, 0, stream>>>(P(M2), m2t, 256, 128);
  }

  // 2) kNN graphs
  k_knn<3><<<N_PTS / 256, 256, 0, stream>>>(pos, idx_pos);
  k_knn<25><<<N_PTS / 256, 256, 0, stream>>>(x, idx_x);

  // 3) init scatter-max outputs to -inf
  k_fill_f32<<<(N_PTS * 64 + 255) / 256, 256, 0, stream>>>(x1, -INFINITY, N_PTS * 64);
  k_fill_f32<<<(N_PTS * 64 + 255) / 256, 256, 0, stream>>>(x2, -INFINITY, N_PTS * 64);
  k_fill_f32<<<(N_PTS * 64 + 255) / 256, 256, 0, stream>>>(x3, -INFINITY, N_PTS * 64);
  k_fill_f32<<<(N_PTS * 64 + 255) / 256, 256, 0, stream>>>(x4, -INFINITY, N_PTS * 64);

  // 4) conv1: full EdgeConv on pos
  k_edge_layer1<3><<<E_full / 256, 256, 0, stream>>>(pos, idx_pos, 1, E_full,
      P(C1A + 0), P(C1A + 1), P(C1A + 2), P(C1A + 3), P(C1A + 4), P(C1A + 5), e1);
  k_edge_layer2_wmma<<<E_full / 64, 128, 0, stream>>>(e1, w2t[0],
      P(C1B + 1), P(C1B + 2), P(C1B + 3), P(C1B + 4), P(C1B + 5), 1, E_full, x1);

  // 5) conv2: dilated EdgeConv on pos
  k_edge_layer1<3><<<E_dil / 256, 256, 0, stream>>>(pos, idx_pos, DILF, E_dil,
      P(C2A + 0), P(C2A + 1), P(C2A + 2), P(C2A + 3), P(C2A + 4), P(C2A + 5), e1);
  k_edge_layer2_wmma<<<E_dil / 64, 128, 0, stream>>>(e1, w2t[1],
      P(C2B + 1), P(C2B + 2), P(C2B + 3), P(C2B + 4), P(C2B + 5), DILF, E_dil, x2);

  // 6) conv3: full EdgeConv on x
  k_edge_layer1<25><<<E_full / 256, 256, 0, stream>>>(x, idx_x, 1, E_full,
      P(C3A + 0), P(C3A + 1), P(C3A + 2), P(C3A + 3), P(C3A + 4), P(C3A + 5), e1);
  k_edge_layer2_wmma<<<E_full / 64, 128, 0, stream>>>(e1, w2t[2],
      P(C3B + 1), P(C3B + 2), P(C3B + 3), P(C3B + 4), P(C3B + 5), 1, E_full, x3);

  // 7) conv4: dilated EdgeConv on x
  k_edge_layer1<25><<<E_dil / 256, 256, 0, stream>>>(x, idx_x, DILF, E_dil,
      P(C4A + 0), P(C4A + 1), P(C4A + 2), P(C4A + 3), P(C4A + 4), P(C4A + 5), e1);
  k_edge_layer2_wmma<<<E_dil / 64, 128, 0, stream>>>(e1, w2t[3],
      P(C4B + 1), P(C4B + 2), P(C4B + 3), P(C4B + 4), P(C4B + 5), DILF, E_dil, x4);

  // 8) concat -> f16
  k_concat4_f16<<<(N_PTS * 256 + 255) / 256, 256, 0, stream>>>(x1, x2, x3, x4, hcat);

  // 9) dense MLP tail (WMMA + TDM-staged A tiles)
  {
    dim3 g1(N_PTS / 64, 1024 / 64);
    k_lin_wmma<<<g1, 128, 0, stream>>>(hcat, lin1t, 256, 1024,
        P(L1 + 1), P(L1 + 2), P(L1 + 3), P(L1 + 4), P(L1 + 5), h1);
    dim3 g2(N_PTS / 64, 256 / 64);
    k_lin_wmma<<<g2, 128, 0, stream>>>(h1, m1t, 1024, 256,
        P(M1 + 1), P(M1 + 2), P(M1 + 3), P(M1 + 4), P(M1 + 5), h2);
    dim3 g3(N_PTS / 64, 128 / 64);
    k_lin_wmma<<<g3, 128, 0, stream>>>(h2, m2t, 256, 128,
        P(M2 + 1), P(M2 + 2), P(M2 + 3), P(M2 + 4), P(M2 + 5), h3);
  }

  // 10) classifier + log_softmax -> d_out (f32, N x 10)
  k_lout_logsoftmax<<<N_PTS / 256, 256, 0, stream>>>(h3, P(LOUT_W), P(LOUT_B), (float*)d_out);
}
